// NGCF_RNN_48825188221326
// MI455X (gfx1250) — compile-verified, matrix operand order checked by reading, not run on hardware
//
#include <hip/hip_runtime.h>
#include <hip/hip_bf16.h>

#define N_USER 120000
#define N_ITEM 60000
#define NTOT   180000
#define D      64
#define NLAYERS 3
#define LDSF   72   // padded LDS row stride in floats (72*4=288 bytes, 16B aligned)

typedef __attribute__((ext_vector_type(16))) __bf16 v16bf;
typedef __attribute__((ext_vector_type(8)))  float  v8f;
typedef __attribute__((ext_vector_type(4)))  float  fv4;   // clang vector: OK for nontemporal builtins

#define WMMA_BF16(A, B, C) \
  __builtin_amdgcn_wmma_f32_16x16x32_bf16(false, (A), false, (B), (short)0, (C), false, false)

// Native device-scope f32 atomic add (GLOBAL_ATOMIC_ADD_F32, no return -> STOREcnt;
// implicit S_WAIT_IDLE at S_ENDPGM guarantees completion).
__device__ __forceinline__ void atomic_add_f32_dev(float* p, float v) {
  asm volatile("global_atomic_add_f32 %0, %1, off scope:SCOPE_DEV"
               :: "v"(p), "v"(v)
               : "memory");
}

// ---- A fragment (16x32 bf16): lane holds row m, khalf in {0,8}; rowp = &X[row][khalf]
// elements 0..7 -> K = kbase+khalf+0..7 ; elements 8..15 -> K = kbase+khalf+16..23
__device__ __forceinline__ v16bf cvt_a_frag(const float* rowp, int kbase) {
  fv4 x0 = *(const fv4*)(rowp + kbase);
  fv4 x1 = *(const fv4*)(rowp + kbase + 4);
  fv4 x2 = *(const fv4*)(rowp + kbase + 16);
  fv4 x3 = *(const fv4*)(rowp + kbase + 20);
  v16bf a;
  a[0]=(__bf16)x0.x; a[1]=(__bf16)x0.y; a[2]=(__bf16)x0.z; a[3]=(__bf16)x0.w;
  a[4]=(__bf16)x1.x; a[5]=(__bf16)x1.y; a[6]=(__bf16)x1.z; a[7]=(__bf16)x1.w;
  a[8]=(__bf16)x2.x; a[9]=(__bf16)x2.y; a[10]=(__bf16)x2.z; a[11]=(__bf16)x2.w;
  a[12]=(__bf16)x3.x; a[13]=(__bf16)x3.y; a[14]=(__bf16)x3.z; a[15]=(__bf16)x3.w;
  return a;
}

// Same, but with non-temporal loads (streamed once-read data; keep L2 for the gather table)
__device__ __forceinline__ v16bf cvt_a_frag_nt(const float* rowp, int kbase) {
  fv4 x0 = __builtin_nontemporal_load((const fv4*)(rowp + kbase));
  fv4 x1 = __builtin_nontemporal_load((const fv4*)(rowp + kbase + 4));
  fv4 x2 = __builtin_nontemporal_load((const fv4*)(rowp + kbase + 16));
  fv4 x3 = __builtin_nontemporal_load((const fv4*)(rowp + kbase + 20));
  v16bf a;
  a[0]=(__bf16)x0.x; a[1]=(__bf16)x0.y; a[2]=(__bf16)x0.z; a[3]=(__bf16)x0.w;
  a[4]=(__bf16)x1.x; a[5]=(__bf16)x1.y; a[6]=(__bf16)x1.z; a[7]=(__bf16)x1.w;
  a[8]=(__bf16)x2.x; a[9]=(__bf16)x2.y; a[10]=(__bf16)x2.z; a[11]=(__bf16)x2.w;
  a[12]=(__bf16)x3.x; a[13]=(__bf16)x3.y; a[14]=(__bf16)x3.z; a[15]=(__bf16)x3.w;
  return a;
}

// ---- A fragment of elementwise product ego*side
__device__ __forceinline__ v16bf cvt_prod_frag(const float* ep, const float* sp, int kbase) {
  fv4 e0 = *(const fv4*)(ep + kbase);      fv4 s0 = *(const fv4*)(sp + kbase);
  fv4 e1 = *(const fv4*)(ep + kbase + 4);  fv4 s1 = *(const fv4*)(sp + kbase + 4);
  fv4 e2 = *(const fv4*)(ep + kbase + 16); fv4 s2 = *(const fv4*)(sp + kbase + 16);
  fv4 e3 = *(const fv4*)(ep + kbase + 20); fv4 s3 = *(const fv4*)(sp + kbase + 20);
  fv4 p0 = e0 * s0, p1 = e1 * s1, p2 = e2 * s2, p3 = e3 * s3;
  v16bf a;
  a[0]=(__bf16)p0.x; a[1]=(__bf16)p0.y; a[2]=(__bf16)p0.z; a[3]=(__bf16)p0.w;
  a[4]=(__bf16)p1.x; a[5]=(__bf16)p1.y; a[6]=(__bf16)p1.z; a[7]=(__bf16)p1.w;
  a[8]=(__bf16)p2.x; a[9]=(__bf16)p2.y; a[10]=(__bf16)p2.z; a[11]=(__bf16)p2.w;
  a[12]=(__bf16)p3.x; a[13]=(__bf16)p3.y; a[14]=(__bf16)p3.z; a[15]=(__bf16)p3.w;
  return a;
}

// ---- B fragment (32x16 bf16, K x N): lane holds col n; W is 64x64 row-major [k][n]
__device__ __forceinline__ v16bf load_b_frag(const __bf16* W, int kbase, int khalf, int n) {
  v16bf b;
#pragma unroll
  for (int p = 0; p < 16; ++p) {
    int k = kbase + khalf + ((p < 8) ? p : (8 + p));   // p>=8 -> 16+(p-8)
    b[p] = W[(k << 6) + n];
  }
  return b;
}

// ======================= kernels =======================

__global__ void prep_weights(const float* W1, const float* W2,
                             const float* Wih, const float* Whh,
                             __bf16* w1b, __bf16* w2b, __bf16* wihT, __bf16* whhT) {
  for (int i = threadIdx.x; i < 3 * 4096; i += blockDim.x) {
    w1b[i] = (__bf16)W1[i];
    w2b[i] = (__bf16)W2[i];
  }
  for (int i = threadIdx.x; i < 4096; i += blockDim.x) {
    int k = i >> 6, n = i & 63;
    wihT[i] = (__bf16)Wih[n * 64 + k];   // B[k][n] = Wih[n][k]  (x @ Wih^T)
    whhT[i] = (__bf16)Whh[n * 64 + k];
  }
}

__global__ void init_embed(const float* __restrict__ ue, const float* __restrict__ ie,
                           float* ego, float* side, float* lay0) {
  size_t i = (size_t)blockIdx.x * blockDim.x + threadIdx.x;   // fv4 index
  const size_t total = (size_t)NTOT * D / 4;
  if (i >= total) return;
  const size_t uq = (size_t)N_USER * D / 4;
  fv4 v = (i < uq) ? ((const fv4*)ue)[i] : ((const fv4*)ie)[i - uq];
  ((fv4*)ego)[i]  = v;
  ((fv4*)side)[i] = v;
  __builtin_nontemporal_store(v, (fv4*)lay0 + i);   // read once much later (RNN)
}

// side[row] += val * ego[col]; 16 lanes per edge, float4 gather + 4 native f32 atomics
__global__ void spmm_scatter(const float* __restrict__ vals, const int* __restrict__ rows,
                             const int* __restrict__ cols, const float* __restrict__ ego,
                             float* side, int E) {
  long long t = (long long)blockIdx.x * blockDim.x + threadIdx.x;
  int e = (int)(t >> 4);
  if (e >= E) return;
  int c = ((int)t & 15) << 2;
  float v = vals[e];
  fv4 x = *(const fv4*)(ego + (size_t)cols[e] * D + c);
  float* dst = side + (size_t)rows[e] * D + c;
  atomic_add_f32_dev(dst + 0, v * x.x);
  atomic_add_f32_dev(dst + 1, v * x.y);
  atomic_add_f32_dev(dst + 2, v * x.z);
  atomic_add_f32_dev(dst + 3, v * x.w);
}

// ego_new = lrelu(side@W1+b1) + lrelu((ego*side)@W2+b2); ego,side <- ego_new; lay_out <- l2norm(ego_new)
__global__ void __launch_bounds__(128)
layer_kernel(float* ego, float* side,
             const __bf16* __restrict__ W1b, const float* __restrict__ b1,
             const __bf16* __restrict__ W2b, const float* __restrict__ b2,
             float* lay_out) {
  __shared__ float sE[16 * LDSF];
  __shared__ float sRn[16];
  const int tid = threadIdx.x;
  const int lane = tid & 31, wave = tid >> 5;
  const int m0 = lane & 15;
  const int khalf = (lane >> 4) << 3;        // 0 or 8
  const int mh = khalf;                      // C/D row offset for upper lane half
  const int n = (wave << 4) + m0;            // global output column 0..63

  v16bf B1k0 = load_b_frag(W1b, 0, khalf, n);
  v16bf B1k1 = load_b_frag(W1b, 32, khalf, n);
  v16bf B2k0 = load_b_frag(W2b, 0, khalf, n);
  v16bf B2k1 = load_b_frag(W2b, 32, khalf, n);
  const float bb1 = b1[n], bb2 = b2[n];

  const int NB = NTOT / 16;
  for (int rb = blockIdx.x; rb < NB; rb += gridDim.x) {
    const size_t rowbase = (size_t)rb * 16;
    const float* srow = side + (rowbase + m0) * D + khalf;
    const float* erow = ego  + (rowbase + m0) * D + khalf;
    v16bf As0 = cvt_a_frag(srow, 0);
    v16bf As1 = cvt_a_frag(srow, 32);
    v16bf Ap0 = cvt_prod_frag(erow, srow, 0);
    v16bf Ap1 = cvt_prod_frag(erow, srow, 32);
    v8f acc1 = {}; v8f acc2 = {};
    acc1 = WMMA_BF16(As0, B1k0, acc1);
    acc1 = WMMA_BF16(As1, B1k1, acc1);
    acc2 = WMMA_BF16(Ap0, B2k0, acc2);
    acc2 = WMMA_BF16(Ap1, B2k1, acc2);
#pragma unroll
    for (int i = 0; i < 8; ++i) {
      float s1 = acc1[i] + bb1; s1 = (s1 > 0.f) ? s1 : 0.01f * s1;
      float s2 = acc2[i] + bb2; s2 = (s2 > 0.f) ? s2 : 0.01f * s2;
      sE[(i + mh) * LDSF + n] = s1 + s2;
    }
    __syncthreads();
    if (tid < 16) {
      float ss = 0.f;
#pragma unroll
      for (int c = 0; c < D; ++c) { float x = sE[tid * LDSF + c]; ss += x * x; }
      sRn[tid] = 1.0f / fmaxf(sqrtf(ss), 1e-12f);
    }
    __syncthreads();
    {
      int r = tid >> 3, c0 = (tid & 7) << 3;
      float rn = sRn[r];
      size_t g = (rowbase + r) * D + c0;
#pragma unroll
      for (int j = 0; j < 8; ++j) {
        float e = sE[r * LDSF + c0 + j];
        ego[g + j]  = e;          // re-read randomly by next scatter: keep cached (RT)
        side[g + j] = e;          // init next layer's (L+I) accumulator
        __builtin_nontemporal_store(e * rn, lay_out + g + j);  // read once later (RNN)
      }
    }
    __syncthreads();
  }
}

// h=0; for t in 0..3: h = tanh(x_t@WihT + bih + h@WhhT + bhh)  (row-local => fully fused)
__global__ void __launch_bounds__(128)
rnn_kernel(const float* __restrict__ layers,
           const __bf16* __restrict__ WihTb, const __bf16* __restrict__ WhhTb,
           const float* __restrict__ bih, const float* __restrict__ bhh, float* hout) {
  __shared__ float sH[16 * LDSF];
  const int tid = threadIdx.x;
  const int lane = tid & 31, wave = tid >> 5;
  const int m0 = lane & 15;
  const int khalf = (lane >> 4) << 3;
  const int mh = khalf;
  const int n = (wave << 4) + m0;

  v16bf Bi0 = load_b_frag(WihTb, 0, khalf, n);
  v16bf Bi1 = load_b_frag(WihTb, 32, khalf, n);
  v16bf Bh0 = load_b_frag(WhhTb, 0, khalf, n);
  v16bf Bh1 = load_b_frag(WhhTb, 32, khalf, n);
  const float bias = bih[n] + bhh[n];

  const int NB = NTOT / 16;
  const size_t slab = (size_t)NTOT * D;
  for (int rb = blockIdx.x; rb < NB; rb += gridDim.x) {
    const size_t rowbase = (size_t)rb * 16;
    for (int i = tid; i < 16 * LDSF; i += 128) sH[i] = 0.f;
    __syncthreads();
#pragma unroll
    for (int t = 0; t < NLAYERS + 1; ++t) {
      const float* xrow = layers + t * slab + (rowbase + m0) * D + khalf;
      v16bf Ax0 = cvt_a_frag_nt(xrow, 0);    // stream once, don't pollute L2
      v16bf Ax1 = cvt_a_frag_nt(xrow, 32);
      const float* hrow = &sH[m0 * LDSF + khalf];
      v16bf Ah0 = cvt_a_frag(hrow, 0);
      v16bf Ah1 = cvt_a_frag(hrow, 32);
      v8f acc = {};
      acc = WMMA_BF16(Ax0, Bi0, acc);
      acc = WMMA_BF16(Ax1, Bi1, acc);
      acc = WMMA_BF16(Ah0, Bh0, acc);
      acc = WMMA_BF16(Ah1, Bh1, acc);
      __syncthreads();                       // everyone done reading old h
#pragma unroll
      for (int i = 0; i < 8; ++i)
        sH[(i + mh) * LDSF + n] = tanhf(acc[i] + bias);
      __syncthreads();
    }
    {
      int r = tid >> 3, c0 = (tid & 7) << 3;
      size_t g = (rowbase + r) * D + c0;
#pragma unroll
      for (int j = 0; j < 8; ++j) hout[g + j] = sH[r * LDSF + c0 + j];
    }
    __syncthreads();
  }
}

__global__ void score_kernel(const float* __restrict__ h, const int* __restrict__ users,
                             const int* __restrict__ pos, const int* __restrict__ neg,
                             float* out, int B) {
  int b = blockIdx.x * blockDim.x + threadIdx.x;
  if (b >= B) return;
  const fv4* u = (const fv4*)(h + (size_t)users[b] * D);
  const fv4* p = (const fv4*)(h + ((size_t)N_USER + pos[b]) * D);
  const fv4* q = (const fv4*)(h + ((size_t)N_USER + neg[b]) * D);
  float ps = 0.f, ns = 0.f;
#pragma unroll
  for (int i = 0; i < D / 4; ++i) {
    fv4 uu = u[i], pp = p[i], qq = q[i];
    ps += uu.x * pp.x + uu.y * pp.y + uu.z * pp.z + uu.w * pp.w;
    ns += uu.x * qq.x + uu.y * qq.y + uu.z * qq.z + uu.w * qq.w;
  }
  out[2 * b]     = ps;
  out[2 * b + 1] = ns;
}

// ======================= launch =======================

extern "C" void kernel_launch(void* const* d_in, const int* in_sizes, int n_in,
                              void* d_out, int out_size, void* d_ws, size_t ws_size,
                              hipStream_t stream) {
  const float* user_emb = (const float*)d_in[0];
  const float* item_emb = (const float*)d_in[1];
  const float* W1   = (const float*)d_in[2];
  const float* b1   = (const float*)d_in[3];
  const float* W2   = (const float*)d_in[4];
  const float* b2   = (const float*)d_in[5];
  const float* Wih  = (const float*)d_in[6];
  const float* Whh  = (const float*)d_in[7];
  const float* bih  = (const float*)d_in[8];
  const float* bhh  = (const float*)d_in[9];
  const float* lap_vals = (const float*)d_in[10];
  const int* lap_rows = (const int*)d_in[11];
  const int* lap_cols = (const int*)d_in[12];
  const int* users     = (const int*)d_in[13];
  const int* pos_items = (const int*)d_in[14];
  const int* neg_items = (const int*)d_in[15];
  const int E = in_sizes[10];
  const int B = in_sizes[13];
  float* out = (float*)d_out;

  char* ws = (char*)d_ws;
  const size_t slabB = (size_t)NTOT * D * sizeof(float);
  const size_t slabE = (size_t)NTOT * D;
  float*  d_ego    = (float*)(ws);
  float*  d_side   = (float*)(ws + slabB);
  float*  d_layers = (float*)(ws + 2 * slabB);   // 4 slabs
  float*  d_h      = (float*)(ws + 6 * slabB);
  __bf16* w1b  = (__bf16*)(ws + 7 * slabB);
  __bf16* w2b  = w1b + 3 * 4096;
  __bf16* wihT = w1b + 6 * 4096;
  __bf16* whhT = w1b + 7 * 4096;

  prep_weights<<<1, 256, 0, stream>>>(W1, W2, Wih, Whh, w1b, w2b, wihT, whhT);

  {
    int total4 = NTOT * D / 4;
    init_embed<<<(total4 + 255) / 256, 256, 0, stream>>>(user_emb, item_emb,
                                                         d_ego, d_side, d_layers);
  }

  for (int l = 0; l < NLAYERS; ++l) {
    long long tScat = (long long)E * 16;
    int gScat = (int)((tScat + 255) / 256);
    spmm_scatter<<<gScat, 256, 0, stream>>>(lap_vals, lap_rows, lap_cols, d_ego, d_side, E);
    layer_kernel<<<2048, 128, 0, stream>>>(d_ego, d_side,
                                           w1b + (size_t)l * 4096, b1 + l * 64,
                                           w2b + (size_t)l * 4096, b2 + l * 64,
                                           d_layers + (size_t)(l + 1) * slabE);
  }

  rnn_kernel<<<2048, 128, 0, stream>>>(d_layers, wihT, whhT, bih, bhh, d_h);

  score_kernel<<<(B + 255) / 256, 256, 0, stream>>>(d_h, users, pos_items, neg_items, out, B);
}
